// QRNN_82514911690928
// MI455X (gfx1250) — matrix operand order repaired and checked by reading, not compiled, not run on hardware
//
#include <hip/hip_runtime.h>
#include <hip/hip_bf16.h>

// Problem constants (match reference)
#define T_STEPS 2048
#define BB 64
#define FF 256
#define HH 256
#define LDSP 264  // padded LDS row stride in elements (528B -> 4-bank rotation per lane)

typedef __attribute__((ext_vector_type(16))) __bf16 v16bf;
typedef __attribute__((ext_vector_type(8)))  __bf16 v8bf;
typedef __attribute__((ext_vector_type(8)))  float  v8f;

#define WMMA_BF16(a, b, c) \
  __builtin_amdgcn_wmma_f32_16x16x32_bf16(false, (a), false, (b), (short)0, (c), false, false)

// A fragment (16x32 bf16): lane m(0-15)/m+16 holds row m, K = koff..koff+7 and koff+16..koff+23
static __device__ __forceinline__ v16bf ldA_bf16(const __bf16* p) {
  v8bf x = *reinterpret_cast<const v8bf*>(p);
  v8bf y = *reinterpret_cast<const v8bf*>(p + 16);
  v16bf r;
#pragma unroll
  for (int i = 0; i < 8; ++i) { r[i] = x[i]; r[i + 8] = y[i]; }
  return r;
}

// B fragment (32x16 bf16): lane n(0-15)/n+16 holds column n, K = koff..koff+7 and koff+16..koff+23
static __device__ __forceinline__ v16bf ldB_bf16(const __bf16* p) {
  v16bf r;
#pragma unroll
  for (int i = 0; i < 8; ++i) { r[i] = p[i * LDSP]; r[i + 8] = p[(i + 16) * LDSP]; }
  return r;
}

// A fragment from f32 global with on-the-fly bf16 conversion
static __device__ __forceinline__ v16bf ldA_f32(const float* p) {
  v16bf r;
#pragma unroll
  for (int i = 0; i < 8; ++i) r[i] = (__bf16)p[i];
#pragma unroll
  for (int i = 0; i < 8; ++i) r[i + 8] = (__bf16)p[i + 16];
  return r;
}

// ---------------------------------------------------------------------------
// Kernel 0: build bf16 weights in workspace:
//   wxWT[f][h]  = wx_W[h][f]         (B-operand layout for the input GEMM)
//   Wq[i][o]    = Hamilton-product matrix from qr,qi,qj,qk (B-operand layout)
//   fcoT[h][f]  = fco_W[f][h]        (B-operand layout for the output GEMM)
// ---------------------------------------------------------------------------
__global__ __launch_bounds__(256) void prep_weights(
    const float* __restrict__ wx_W, const float* __restrict__ qr,
    const float* __restrict__ qi, const float* __restrict__ qj,
    const float* __restrict__ qk, const float* __restrict__ fco_W,
    __bf16* __restrict__ wxWT, __bf16* __restrict__ Wq, __bf16* __restrict__ fcoT) {
  int idx = blockIdx.x * blockDim.x + threadIdx.x;
  if (idx >= HH * FF) return;
  int row = idx >> 8, col = idx & 255;
  wxWT[idx] = (__bf16)wx_W[col * FF + row];   // row=f, col=h
  fcoT[idx] = (__bf16)fco_W[col * HH + row];  // row=h, col=f
  // Quaternion weight: row=i (input), col=o (output)
  int br = row >> 6, a = row & 63;
  int bc = col >> 6, b = col & 63;
  const float* q[4] = {qr, qi, qj, qk};
  const int   sel[4][4] = {{0, 1, 2, 3}, {1, 0, 3, 2}, {2, 3, 0, 1}, {3, 2, 1, 0}};
  const float sgn[4][4] = {{1.f, -1.f, -1.f, -1.f},
                           {1.f,  1.f, -1.f,  1.f},
                           {1.f,  1.f,  1.f, -1.f},
                           {1.f, -1.f,  1.f,  1.f}};
  Wq[idx] = (__bf16)(sgn[bc][br] * q[sel[bc][br]][a * 64 + b]);
}

// ---------------------------------------------------------------------------
// Kernel 1: seq[tb][h] = bf16( x[tb][:] @ wxWT + wx_b )   (chip-wide GEMM)
// 256 threads = 8 waves; each wave owns one 16-row strip x all 16 N-tiles.
// ---------------------------------------------------------------------------
__global__ __launch_bounds__(256) void gemm_in(
    const float* __restrict__ x, const __bf16* __restrict__ wxWT,
    const float* __restrict__ wx_b, __bf16* __restrict__ seq) {
  extern __shared__ unsigned char smem_raw[];
  __bf16* sW = reinterpret_cast<__bf16*>(smem_raw);  // [FF][LDSP]
  for (int idx = threadIdx.x; idx < FF * HH; idx += blockDim.x) {
    int k = idx >> 8, n = idx & 255;
    sW[k * LDSP + n] = wxWT[idx];
  }
  __syncthreads();

  const int lane  = threadIdx.x & 31;
  const int wave  = threadIdx.x >> 5;
  const int m0    = (blockIdx.x * 8 + wave) * 16;
  const int mrow  = lane & 15;
  const int koff  = (lane >> 4) << 3;
  const int nlane = lane & 15;
  const int mh    = (lane >> 4) << 3;

  v8f acc[16];
#pragma unroll
  for (int nt = 0; nt < 16; ++nt) {
    float bias = wx_b[nt * 16 + nlane];
#pragma unroll
    for (int r = 0; r < 8; ++r) acc[nt][r] = bias;
  }

  const float* aptr = x + (size_t)(m0 + mrow) * FF + koff;
#pragma unroll
  for (int k0 = 0; k0 < FF; k0 += 32) {
    v16bf a = ldA_f32(aptr + k0);
    const __bf16* bbase = sW + (k0 + koff) * LDSP + nlane;
#pragma unroll
    for (int nt = 0; nt < 16; ++nt) {
      v16bf b = ldB_bf16(bbase + nt * 16);
      acc[nt] = WMMA_BF16(a, b, acc[nt]);
    }
  }

  __bf16* obase = seq + (size_t)(m0 + mh) * HH + nlane;
#pragma unroll
  for (int nt = 0; nt < 16; ++nt)
#pragma unroll
    for (int r = 0; r < 8; ++r)
      obase[(size_t)r * HH + nt * 16] = (__bf16)acc[nt][r];
}

// ---------------------------------------------------------------------------
// Kernel 2: the serial recurrence, single workgroup (1 WGP), 32 waves.
// Phase 1 (2048 steps): h <- seq[t] + h@Wq + qb        (consumes seq)
// Phase 2 (2048 steps): h <- h@Wq + qb ; seq[t] = h    (produces seq = hseq)
// Wq and double-buffered bf16 h-state live in LDS; one barrier per step.
// ---------------------------------------------------------------------------
__global__ __launch_bounds__(1024) void recurrence(
    const __bf16* __restrict__ Wq, const float* __restrict__ qb,
    __bf16* __restrict__ seq) {
  extern __shared__ unsigned char smem_raw[];
  __bf16* sQ = reinterpret_cast<__bf16*>(smem_raw);                       // [HH][LDSP]
  __bf16* hb = reinterpret_cast<__bf16*>(smem_raw + HH * LDSP * 2);      // [2][BB][LDSP]

  for (int idx = threadIdx.x; idx < HH * HH; idx += blockDim.x) {
    int k = idx >> 8, n = idx & 255;
    sQ[k * LDSP + n] = Wq[idx];
  }
  for (int idx = threadIdx.x; idx < 2 * BB * LDSP; idx += blockDim.x)
    hb[idx] = (__bf16)0.f;
  __syncthreads();

  const int lane  = threadIdx.x & 31;
  const int wave  = threadIdx.x >> 5;       // 32 waves -> 64 tiles, 2 per wave
  const int mt    = wave & 3;               // 4 M-tiles (64 rows)
  const int ntA   = (wave >> 2) * 2;        // 16 N-tiles, pairs
  const int ntB   = ntA + 1;
  const int m0    = mt * 16;
  const int mrow  = lane & 15;
  const int koff  = (lane >> 4) << 3;
  const int nlane = lane & 15;
  const int mh    = (lane >> 4) << 3;
  const float biasA = qb[ntA * 16 + nlane];
  const float biasB = qb[ntB * 16 + nlane];

  int cur = 0;

  // ---- Phase 1 ----
  for (int t = 0; t < T_STEPS; ++t) {
    const __bf16* wt = seq + (size_t)t * BB * HH + (size_t)(m0 + mh) * HH + nlane;
    v8f accA, accB;
#pragma unroll
    for (int r = 0; r < 8; ++r) {
      accA[r] = biasA + (float)wt[(size_t)r * HH + ntA * 16];
      accB[r] = biasB + (float)wt[(size_t)r * HH + ntB * 16];
    }
    const __bf16* hbase = hb + cur * (BB * LDSP) + (m0 + mrow) * LDSP + koff;
    const __bf16* qbase = sQ + koff * LDSP + nlane;
#pragma unroll
    for (int k0 = 0; k0 < HH; k0 += 32) {
      v16bf a  = ldA_bf16(hbase + k0);
      v16bf b0 = ldB_bf16(qbase + k0 * LDSP + ntA * 16);
      accA = WMMA_BF16(a, b0, accA);
      v16bf b1 = ldB_bf16(qbase + k0 * LDSP + ntB * 16);
      accB = WMMA_BF16(a, b1, accB);
    }
    __bf16* hout = hb + (cur ^ 1) * (BB * LDSP) + (m0 + mh) * LDSP + nlane;
#pragma unroll
    for (int r = 0; r < 8; ++r) {
      hout[r * LDSP + ntA * 16] = (__bf16)accA[r];
      hout[r * LDSP + ntB * 16] = (__bf16)accB[r];
    }
    __syncthreads();
    cur ^= 1;
  }

  // ---- Phase 2 ----
  for (int t = 0; t < T_STEPS; ++t) {
    v8f accA, accB;
#pragma unroll
    for (int r = 0; r < 8; ++r) { accA[r] = biasA; accB[r] = biasB; }
    const __bf16* hbase = hb + cur * (BB * LDSP) + (m0 + mrow) * LDSP + koff;
    const __bf16* qbase = sQ + koff * LDSP + nlane;
#pragma unroll
    for (int k0 = 0; k0 < HH; k0 += 32) {
      v16bf a  = ldA_bf16(hbase + k0);
      v16bf b0 = ldB_bf16(qbase + k0 * LDSP + ntA * 16);
      accA = WMMA_BF16(a, b0, accA);
      v16bf b1 = ldB_bf16(qbase + k0 * LDSP + ntB * 16);
      accB = WMMA_BF16(a, b1, accB);
    }
    __bf16* hout = hb + (cur ^ 1) * (BB * LDSP) + (m0 + mh) * LDSP + nlane;
    __bf16* gout = seq + (size_t)t * BB * HH + (size_t)(m0 + mh) * HH + nlane;
#pragma unroll
    for (int r = 0; r < 8; ++r) {
      __bf16 va = (__bf16)accA[r];
      __bf16 vb = (__bf16)accB[r];
      hout[r * LDSP + ntA * 16] = va;
      hout[r * LDSP + ntB * 16] = vb;
      gout[(size_t)r * HH + ntA * 16] = va;
      gout[(size_t)r * HH + ntB * 16] = vb;
    }
    __syncthreads();
    cur ^= 1;
  }
}

// ---------------------------------------------------------------------------
// Kernel 3: out[tb][f] = sigmoid( hseq[tb][:] @ fcoT + fco_b )  (chip-wide)
// ---------------------------------------------------------------------------
__global__ __launch_bounds__(256) void gemm_out_sig(
    const __bf16* __restrict__ seq, const __bf16* __restrict__ fcoT,
    const float* __restrict__ fco_b, float* __restrict__ out) {
  extern __shared__ unsigned char smem_raw[];
  __bf16* sW = reinterpret_cast<__bf16*>(smem_raw);  // [HH][LDSP]
  for (int idx = threadIdx.x; idx < HH * FF; idx += blockDim.x) {
    int k = idx >> 8, n = idx & 255;
    sW[k * LDSP + n] = fcoT[idx];
  }
  __syncthreads();

  const int lane  = threadIdx.x & 31;
  const int wave  = threadIdx.x >> 5;
  const int m0    = (blockIdx.x * 8 + wave) * 16;
  const int mrow  = lane & 15;
  const int koff  = (lane >> 4) << 3;
  const int nlane = lane & 15;
  const int mh    = (lane >> 4) << 3;

  v8f acc[16];
#pragma unroll
  for (int nt = 0; nt < 16; ++nt) {
    float bias = fco_b[nt * 16 + nlane];
#pragma unroll
    for (int r = 0; r < 8; ++r) acc[nt][r] = bias;
  }

  const __bf16* aptr = seq + (size_t)(m0 + mrow) * HH + koff;
#pragma unroll
  for (int k0 = 0; k0 < HH; k0 += 32) {
    v16bf a = ldA_bf16(aptr + k0);
    const __bf16* bbase = sW + (k0 + koff) * LDSP + nlane;
#pragma unroll
    for (int nt = 0; nt < 16; ++nt) {
      v16bf b = ldB_bf16(bbase + nt * 16);
      acc[nt] = WMMA_BF16(a, b, acc[nt]);
    }
  }

  float* obase = out + (size_t)(m0 + mh) * FF + nlane;
#pragma unroll
  for (int nt = 0; nt < 16; ++nt)
#pragma unroll
    for (int r = 0; r < 8; ++r) {
      float v = acc[nt][r];
      obase[(size_t)r * FF + nt * 16] = 1.0f / (1.0f + __expf(-v));
    }
}

// ---------------------------------------------------------------------------
extern "C" void kernel_launch(void* const* d_in, const int* in_sizes, int n_in,
                              void* d_out, int out_size, void* d_ws, size_t ws_size,
                              hipStream_t stream) {
  const float* x     = (const float*)d_in[0];
  const float* wx_W  = (const float*)d_in[1];
  const float* wx_b  = (const float*)d_in[2];
  const float* qr    = (const float*)d_in[3];
  const float* qi    = (const float*)d_in[4];
  const float* qj    = (const float*)d_in[5];
  const float* qk    = (const float*)d_in[6];
  const float* qb    = (const float*)d_in[7];
  const float* fco_W = (const float*)d_in[8];
  const float* fco_b = (const float*)d_in[9];
  float* out = (float*)d_out;

  // Workspace layout (bf16): [seq (T*B*H)] [wxWT (F*H)] [Wq (H*H)] [fcoT (H*F)]
  const size_t seqElems = (size_t)T_STEPS * BB * HH;  // 33,554,432 (64 MiB)
  __bf16* seq  = (__bf16*)d_ws;
  __bf16* wxWT = seq + seqElems;
  __bf16* Wq   = wxWT + (size_t)FF * HH;
  __bf16* fcoT = Wq + (size_t)HH * HH;

  const size_t ldsGemm = (size_t)FF * LDSP * sizeof(__bf16);                      // 132 KB
  const size_t ldsRec  = (size_t)HH * LDSP * sizeof(__bf16)
                       + (size_t)2 * BB * LDSP * sizeof(__bf16);                  // ~198 KB (<320 KB WGP LDS)

  prep_weights<<<(HH * FF + 255) / 256, 256, 0, stream>>>(
      wx_W, qr, qi, qj, qk, fco_W, wxWT, Wq, fcoT);

  // (T*B)/(8 waves * 16 rows) = 1024 blocks
  gemm_in<<<1024, 256, ldsGemm, stream>>>(x, wxWT, wx_b, seq);

  recurrence<<<1, 1024, ldsRec, stream>>>(Wq, qb, seq);

  gemm_out_sig<<<1024, 256, ldsGemm, stream>>>(seq, fcoT, fco_b, out);
}
